// LSTMNet_18116172055220
// MI455X (gfx1250) — matrix operand order repaired and checked by reading, not compiled, not run on hardware
//
#include <hip/hip_runtime.h>
#include <hip/hip_bf16.h>

typedef __attribute__((ext_vector_type(16))) _Float16 v16h;
typedef __attribute__((ext_vector_type(8)))  _Float16 v8h;
typedef __attribute__((ext_vector_type(4)))  _Float16 v4h;
typedef __attribute__((ext_vector_type(8)))  float    v8f;

#define N_HID   128
#define K_CAT   256
#define BATCH   131072
#define LDW     264   // W4 LDS row stride (f16): 528B, rotates banks lane-to-lane
#define LDA     264   // A-tile stride
#define LDH     136   // hidden-tile stride

__device__ __forceinline__ float sigmoid_(float x) {
    return 1.0f / (1.0f + __expf(-x));
}
__device__ __forceinline__ float tanh_(float x) {
    float e = __expf(2.0f * x);
    return (e - 1.0f) / (e + 1.0f);
}

// 16 consecutive f32 from global -> v16h (register B-fragment for Wfc)
__device__ __forceinline__ v16h load_w_frag_g(const float* __restrict__ p) {
    v16h r;
    const float4* q = (const float4*)p;
#pragma unroll
    for (int i = 0; i < 4; ++i) {
        float4 v = q[i];
        r[4*i+0] = (_Float16)v.x; r[4*i+1] = (_Float16)v.y;
        r[4*i+2] = (_Float16)v.z; r[4*i+3] = (_Float16)v.w;
    }
    return r;
}

// B-fragment from LDS: 16 contiguous f16 (K-elements of one weight row)
__device__ __forceinline__ v16h load_b_frag(const _Float16* __restrict__ p) {
    v8h lo = *(const v8h*)p;
    v8h hi = *(const v8h*)(p + 8);
    v16h r;
#pragma unroll
    for (int i = 0; i < 8; ++i) { r[i] = lo[i]; r[8+i] = hi[i]; }
    return r;
}

// A-fragment from LDS: elems 0..15 = K kb..kb+15 (contiguous)
__device__ __forceinline__ v16h load_a_frag(const _Float16* __restrict__ lds,
                                            int row, int kb, int ld) {
    const _Float16* p = lds + row * ld + kb;
    v8h lo = *(const v8h*)p;
    v8h hi = *(const v8h*)(p + 8);
    v16h r;
#pragma unroll
    for (int i = 0; i < 8; ++i) { r[i] = lo[i]; r[8+i] = hi[i]; }
    return r;
}

__global__ __launch_bounds__(256, 1)
void lstm_fused_wmma(const float* __restrict__ x,  const float* __restrict__ hidden,
                     const float* __restrict__ ctx,
                     const float* __restrict__ Wf, const float* __restrict__ bf,
                     const float* __restrict__ Wi, const float* __restrict__ bi,
                     const float* __restrict__ Wc, const float* __restrict__ bc,
                     const float* __restrict__ Wo, const float* __restrict__ bo,
                     const float* __restrict__ Wfc,const float* __restrict__ bfc,
                     float* __restrict__ out, float* __restrict__ hid_out,
                     float* __restrict__ ctx_out, int tiles_per_block)
{
    __shared__ _Float16 lds_w[512 * LDW]; // all 4 gate weight matrices, f16 (~270KB)
    __shared__ _Float16 lds_a[16 * LDA];  // cat(hidden,x) tile
    __shared__ _Float16 lds_h[16 * LDH];  // hidden_new tile

    const int tid  = threadIdx.x;
    const int lane = tid & 31;
    const int wave = tid >> 5;        // 0..7: hidden-column slice j
    const int ln   = lane & 15;
    const int hi   = lane >> 4;       // lane half (K-split select)
    const int ncol = wave * 16 + ln;  // 0..127: this lane's output column

    // ---- stage W4 = [Wf;Wi;Wc;Wo] into LDS as f16, once per block ----
    {
        const float* Wg[4] = {Wf, Wi, Wc, Wo};
#pragma unroll
        for (int g = 0; g < 4; ++g) {
            const float4* W = (const float4*)Wg[g]; // [128,256] = 8192 float4
#pragma unroll 4
            for (int i = 0; i < 32; ++i) {
                int f   = tid + i * 256;  // 0..8191
                int row = f >> 6;         // 64 float4 per row
                int c4  = f & 63;
                float4 v = W[f];
                v4h h;
                h[0] = (_Float16)v.x; h[1] = (_Float16)v.y;
                h[2] = (_Float16)v.z; h[3] = (_Float16)v.w;
                *(v4h*)&lds_w[(size_t)(g * 128 + row) * LDW + c4 * 4] = h;
            }
        }
    }

    // ---- Wfc stays register-stationary (only 32 VGPRs) ----
    v16h bwfc[4];
    {
        const float* wrow = Wfc + (size_t)ncol * N_HID;
#pragma unroll
        for (int s = 0; s < 4; ++s)
            bwfc[s] = load_w_frag_g(wrow + s * 32 + hi * 16);
    }
    const float bfv = bf[ncol], biv = bi[ncol], bcv = bc[ncol], bov = bo[ncol];
    const float bfcv = bfc[ncol];

    // per-gate LDS base pointers for this lane's B-fragments
    const _Float16* wb0 = lds_w + (size_t)(0 * 128 + ncol) * LDW + hi * 16;
    const _Float16* wb1 = lds_w + (size_t)(1 * 128 + ncol) * LDW + hi * 16;
    const _Float16* wb2 = lds_w + (size_t)(2 * 128 + ncol) * LDW + hi * 16;
    const _Float16* wb3 = lds_w + (size_t)(3 * 128 + ncol) * LDW + hi * 16;

    for (int t = 0; t < tiles_per_block; ++t) {
        const int row0 = (blockIdx.x * tiles_per_block + t) * 16;
        __syncthreads();  // previous iteration's LDS readers done (also covers W4 staging)

        // ---- stage cat(hidden, x) rows [16 x 256] -> LDS f16 (coalesced) ----
#pragma unroll
        for (int i = 0; i < 16; ++i) {
            int e = tid + i * 256;          // 0..4095
            int r = e >> 8, c = e & 255;
            float v = (c < 128) ? hidden[(size_t)(row0 + r) * 128 + c]
                                : x[(size_t)(row0 + r) * 128 + (c - 128)];
            lds_a[r * LDA + c] = (_Float16)v;
        }
        __syncthreads();

        // ---- A fragments (shared by all 4 gate WMMAs) ----
        v16h af[8];
#pragma unroll
        for (int s = 0; s < 8; ++s)
            af[s] = load_a_frag(lds_a, ln, s * 32 + hi * 16, LDA);

        // ---- GEMM1: gates[16 x 16] x 4, K=256, B-fragments streamed from LDS ----
        v8f accf, acci, accc, acco;
#pragma unroll
        for (int r = 0; r < 8; ++r) { accf[r] = bfv; acci[r] = biv; accc[r] = bcv; acco[r] = bov; }
#pragma unroll
        for (int s = 0; s < 8; ++s) {
            v16h b0 = load_b_frag(wb0 + s * 32);
            v16h b1 = load_b_frag(wb1 + s * 32);
            v16h b2 = load_b_frag(wb2 + s * 32);
            v16h b3 = load_b_frag(wb3 + s * 32);
            accf = __builtin_amdgcn_wmma_f32_16x16x32_f16(false, af[s], false, b0, (short)0, accf, false, false);
            acci = __builtin_amdgcn_wmma_f32_16x16x32_f16(false, af[s], false, b1, (short)0, acci, false, false);
            accc = __builtin_amdgcn_wmma_f32_16x16x32_f16(false, af[s], false, b2, (short)0, accc, false, false);
            acco = __builtin_amdgcn_wmma_f32_16x16x32_f16(false, af[s], false, b3, (short)0, acco, false, false);
        }

        // ---- pointwise LSTM update; write ctx/hidden; stage hidden f16 ----
#pragma unroll
        for (int r = 0; r < 8; ++r) {
            int row = row0 + r + hi * 8;                 // C-layout: vgpr r -> row
            size_t idx = (size_t)row * 128 + ncol;
            float f_t = sigmoid_(accf[r]);
            float i_t = sigmoid_(acci[r]);
            float c_t = tanh_(accc[r]);
            float o_t = sigmoid_(acco[r]);
            float cn  = f_t * ctx[idx] + i_t * c_t;
            float hn  = o_t * tanh_(cn);
            ctx_out[idx] = cn;
            hid_out[idx] = hn;
            lds_h[(r + hi * 8) * LDH + ncol] = (_Float16)hn;
        }
        __syncthreads();

        // ---- GEMM2: out[16 x 16] slice, K=128, Wfc from registers ----
        v16h ah[4];
#pragma unroll
        for (int s = 0; s < 4; ++s)
            ah[s] = load_a_frag(lds_h, ln, s * 32 + hi * 16, LDH);
        v8f acc;
#pragma unroll
        for (int r = 0; r < 8; ++r) acc[r] = bfcv;
#pragma unroll
        for (int s = 0; s < 4; ++s)
            acc = __builtin_amdgcn_wmma_f32_16x16x32_f16(false, ah[s], false, bwfc[s], (short)0, acc, false, false);
#pragma unroll
        for (int r = 0; r < 8; ++r) {
            int row = row0 + r + hi * 8;
            out[(size_t)row * 128 + ncol] = acc[r];
        }
    }
}

extern "C" void kernel_launch(void* const* d_in, const int* in_sizes, int n_in,
                              void* d_out, int out_size, void* d_ws, size_t ws_size,
                              hipStream_t stream) {
    (void)in_sizes; (void)n_in; (void)out_size; (void)d_ws; (void)ws_size;
    const float* x      = (const float*)d_in[0];
    const float* hidden = (const float*)d_in[1];
    const float* ctx    = (const float*)d_in[2];
    const float* Wf     = (const float*)d_in[3];
    const float* bf     = (const float*)d_in[4];
    const float* Wi     = (const float*)d_in[5];
    const float* bi     = (const float*)d_in[6];
    const float* Wc     = (const float*)d_in[7];
    const float* bc     = (const float*)d_in[8];
    const float* Wo     = (const float*)d_in[9];
    const float* bo     = (const float*)d_in[10];
    const float* Wfc    = (const float*)d_in[11];
    const float* bfc    = (const float*)d_in[12];

    float* out     = (float*)d_out;                           // [B,128]
    float* hid_out = (float*)d_out + (size_t)BATCH * 128;     // [B,128]
    float* ctx_out = (float*)d_out + (size_t)2 * BATCH * 128; // [B,128]

    const int grid = 1024;                            // 128 rows per block
    const int tiles_per_block = BATCH / (16 * grid);  // 8 row-tiles of 16
    lstm_fused_wmma<<<grid, 256, 0, stream>>>(x, hidden, ctx,
                                              Wf, bf, Wi, bi, Wc, bc, Wo, bo,
                                              Wfc, bfc,
                                              out, hid_out, ctx_out,
                                              tiles_per_block);
}
